// Model_33406255628698
// MI455X (gfx1250) — compile-verified, matrix operand order checked by reading, not run on hardware
//
#include <hip/hip_runtime.h>
#include <hip/hip_bf16.h>

// ---------------------------------------------------------------------------
// DialogueRNN-style model on MI455X (gfx1250).
// bf16 WMMA GEMMs with async global->LDS tile staging + f32 epilogues.
// ---------------------------------------------------------------------------

typedef __attribute__((ext_vector_type(16))) __bf16 v16bf;
typedef __attribute__((ext_vector_type(8)))  __bf16 v8bf;
typedef __attribute__((ext_vector_type(8)))  float  v8f;

#define B_DLG    32
#define T_UTT    64
#define TW_WORDS 32
#define EW_IN    300
#define EW_PAD   320
#define H_LSTM   512
#define G4H      2048   // 4*H
#define E_UTT    1024
#define E3       3072   // 3*E
#define HH_EMO   512
#define HH3      1536   // 3*HH
#define C_CLS    7
#define N_UTTER  2048   // B*T

// ---------------------------------------------------------------------------
// CDNA5 async copy: global -> LDS, 16 bytes, tracked by ASYNCcnt.
// Low 32 bits of a generic pointer to __shared__ are the raw LDS offset
// (flat-aperture rule: LDS address is the low 32 bits).
// ---------------------------------------------------------------------------
__device__ __forceinline__ void async_copy_b128(void* lds_ptr, const void* gptr) {
    uint32_t l = (uint32_t)(uintptr_t)lds_ptr;
    asm volatile("global_load_async_to_lds_b128 %0, %1, off"
                 :: "v"(l), "v"(gptr) : "memory");
}
__device__ __forceinline__ void wait_async() {
    asm volatile("s_wait_asynccnt 0x0" ::: "memory");
}

// Build a 16x32 bf16 WMMA operand fragment from an LDS row (row-major,
// 32 bf16 per row). CDNA5 16-bit operand layout:
//   lane<16  -> K {half*8=0..7, 16..23};  lane>=16 -> K {8..15, 24..31}
// p points at (row, half*8); second 16B chunk is +16 elements.
__device__ __forceinline__ v16bf frag_from_lds(const __bf16* p) {
    v8bf lo = *(const v8bf*)(p);
    v8bf hi = *(const v8bf*)(p + 16);
    v16bf f;
#pragma unroll
    for (int i = 0; i < 8; ++i) { f[i] = lo[i]; f[i + 8] = hi[i]; }
    return f;
}

// ---------------------------------------------------------------------------
// One GEMM segment: acc += A[M,K] * Bw[N,K]^T over the block tile.
// Block tile 128(M) x 64(N), K step 32. LDS: sA 128x32, sB 64x32 (bf16).
// Rows are CLAMPED (not predicated): garbage rows only feed C rows/cols that
// the store epilogue masks out, so the hot loop stays divergence-free.
// ---------------------------------------------------------------------------
__device__ __forceinline__ void gemm_segment(
    const __bf16* __restrict__ A, int lda,
    const __bf16* __restrict__ Bw, int ldb, int K, int M, int N,
    int bm, int bn, int tid, int lane, int wm, int wn,
    __bf16* sA, __bf16* sB, v8f acc[2][2])
{
    const int half = lane >> 4;
    const int rr   = lane & 15;
    const __bf16* pa0 = sA + (wm * 32 + rr) * 32 + half * 8;
    const __bf16* pa1 = pa0 + 16 * 32;
    const __bf16* pb0 = sB + (wn * 32 + rr) * 32 + half * 8;
    const __bf16* pb1 = pb0 + 16 * 32;

    // cooperative-load chunk mapping (16B chunks of a 32-element row):
    const int arow0 = tid >> 2;              // A chunk 0: rows 0..63
    const int arow1 = (tid + 256) >> 2;      // A chunk 1: rows 64..127
    const int acol  = (tid & 3) * 8;
    const int brow  = tid >> 2;              // B: rows 0..63
    const int bcol  = (tid & 3) * 8;
    int ga0 = bm + arow0; ga0 = ga0 < M ? ga0 : M - 1;
    int ga1 = bm + arow1; ga1 = ga1 < M ? ga1 : M - 1;
    int gb  = bn + brow;  gb  = gb  < N ? gb  : N - 1;
    const __bf16* gA0 = A  + (size_t)ga0 * lda + acol;
    const __bf16* gA1 = A  + (size_t)ga1 * lda + acol;
    const __bf16* gB  = Bw + (size_t)gb  * ldb + bcol;
    __bf16* lA0 = sA + arow0 * 32 + acol;
    __bf16* lA1 = sA + arow1 * 32 + acol;
    __bf16* lB  = sB + brow  * 32 + bcol;

    for (int k = 0; k < K; k += 32) {
        __syncthreads();                      // LDS safe to overwrite
        async_copy_b128(lA0, gA0 + k);
        async_copy_b128(lA1, gA1 + k);
        async_copy_b128(lB,  gB  + k);
        __builtin_prefetch(gA0 + k + 32, 0, 1);   // next A K-tile -> L2
        wait_async();
        __syncthreads();                      // all waves' tiles resident

        v16bf a0 = frag_from_lds(pa0);
        v16bf a1 = frag_from_lds(pa1);
        v16bf b0 = frag_from_lds(pb0);
        v16bf b1 = frag_from_lds(pb1);
        acc[0][0] = __builtin_amdgcn_wmma_f32_16x16x32_bf16(false, a0, false, b0, (short)0, acc[0][0], false, false);
        acc[0][1] = __builtin_amdgcn_wmma_f32_16x16x32_bf16(false, a0, false, b1, (short)0, acc[0][1], false, false);
        acc[1][0] = __builtin_amdgcn_wmma_f32_16x16x32_bf16(false, a1, false, b0, (short)0, acc[1][0], false, false);
        acc[1][1] = __builtin_amdgcn_wmma_f32_16x16x32_bf16(false, a1, false, b1, (short)0, acc[1][1], false, false);
    }
}

// ---------------------------------------------------------------------------
// General two-segment bf16 WMMA GEMM:
//   C[M,N] = A1[M,K1]*B1[N,K1]^T (+ A2[M,K2]*B2[N,K2]^T) + bias[N]
// Block = 256 threads = 8 waves (4 along M x 2 along N); wave tile 32x32.
// K1/K2 must be multiples of 32 (all buffers zero-padded in K).
// ---------------------------------------------------------------------------
__global__ __launch_bounds__(256) void gemm_bf16_wmma(
    const __bf16* __restrict__ A1, int lda1, const __bf16* __restrict__ B1, int ldb1, int K1,
    const __bf16* __restrict__ A2, int lda2, const __bf16* __restrict__ B2, int ldb2, int K2,
    const float* __restrict__ bias, float* __restrict__ C, int ldc, int M, int N)
{
    __shared__ __bf16 sA[128 * 32];   // 8 KB
    __shared__ __bf16 sB[64 * 32];    // 4 KB

    const int tid  = threadIdx.x;
    const int lane = tid & 31;
    const int wave = tid >> 5;
    const int wm   = wave & 3;
    const int wn   = wave >> 2;
    const int bm   = blockIdx.y * 128;
    const int bn   = blockIdx.x * 64;

    v8f acc[2][2];
#pragma unroll
    for (int i = 0; i < 2; ++i)
#pragma unroll
        for (int j = 0; j < 2; ++j)
#pragma unroll
            for (int v = 0; v < 8; ++v) acc[i][j][v] = 0.0f;

    gemm_segment(A1, lda1, B1, ldb1, K1, M, N, bm, bn, tid, lane, wm, wn, sA, sB, acc);
    if (A2)
        gemm_segment(A2, lda2, B2, ldb2, K2, M, N, bm, bn, tid, lane, wm, wn, sA, sB, acc);

    // C/D layout: lane 0-15 -> N=lane, M=vgpr; lane 16-31 -> M=vgpr+8
#pragma unroll
    for (int i = 0; i < 2; ++i)
#pragma unroll
        for (int j = 0; j < 2; ++j) {
            const int n  = bn + wn * 32 + j * 16 + (lane & 15);
            const int mb = bm + wm * 32 + i * 16 + (lane >> 4) * 8;
            if (n < N) {
                const float bv = bias ? bias[n] : 0.0f;
#pragma unroll
                for (int v = 0; v < 8; ++v) {
                    const int m = mb + v;
                    if (m < M) C[(size_t)m * ldc + n] = acc[i][j][v] + bv;
                }
            }
        }
}

// ---------------------------------------------------------------------------
// f32 -> bf16 conversion with zero K-padding (row-major [R,Cc] -> [R,Cp])
// ---------------------------------------------------------------------------
__global__ void cvt_pad_kernel(const float* __restrict__ src, __bf16* __restrict__ dst,
                               int R, int Cc, int Cp) {
    long idx = (long)blockIdx.x * blockDim.x + threadIdx.x;
    long total = (long)R * Cp;
    if (idx >= total) return;
    int r = (int)(idx / Cp), c = (int)(idx % Cp);
    dst[idx] = (c < Cc) ? (__bf16)src[(size_t)r * Cc + c] : (__bf16)0.0f;
}

// f32 [R,C] -> bf16 transposed [C,R] (W_att is used un-transposed in the ref)
__global__ void cvt_transpose_kernel(const float* __restrict__ src, __bf16* __restrict__ dst,
                                     int R, int Ccols) {
    long idx = (long)blockIdx.x * blockDim.x + threadIdx.x;
    long total = (long)R * Ccols;
    if (idx >= total) return;
    int r = (int)(idx / Ccols), c = (int)(idx % Ccols);
    dst[(size_t)c * R + r] = (__bf16)src[idx];
}

// ---------------------------------------------------------------------------
// LSTM gate epilogue: G[N,4H] (i,f,g,o); c' = sig(f)*c + sig(i)*tanh(g);
// h' = sig(o)*tanh(c'). Writes bf16 h (next GEMM operand) + optional f32 h.
// ---------------------------------------------------------------------------
__global__ void lstm_gate_kernel(const float* __restrict__ G, const float* __restrict__ c_in,
                                 float* __restrict__ c_out, __bf16* __restrict__ h_bf,
                                 float* __restrict__ h_f, int ld_hf, int total) {
    int idx = blockIdx.x * blockDim.x + threadIdx.x;
    if (idx >= total) return;
    int n = idx >> 9;        // / 512
    int j = idx & 511;
    const float* g = G + (size_t)n * G4H;
    float i_ = 1.0f / (1.0f + expf(-g[j]));
    float f_ = 1.0f / (1.0f + expf(-g[512 + j]));
    float gg = tanhf(g[1024 + j]);
    float o_ = 1.0f / (1.0f + expf(-g[1536 + j]));
    float c2 = f_ * c_in[idx] + i_ * gg;
    float h2 = o_ * tanhf(c2);
    c_out[idx] = c2;
    h_bf[idx]  = (__bf16)h2;
    if (h_f) h_f[(size_t)n * ld_hf + j] = h2;
}

// ---------------------------------------------------------------------------
// GRU gate epilogue (gate order r,z,n):
//   r=sig(gi_r+gh_r); z=sig(gi_z+gh_z); n=tanh(gi_n + r*gh_n)
//   h' = (1-z)*n + z*h
// ---------------------------------------------------------------------------
__global__ void gru_gate_kernel(const float* __restrict__ gi, const float* __restrict__ gh,
                                const float* __restrict__ h, float* __restrict__ h_new,
                                __bf16* __restrict__ h_bf, __bf16* __restrict__ h_bf2,
                                int ld_bf2, int Gsz, int total) {
    int idx = blockIdx.x * blockDim.x + threadIdx.x;
    if (idx >= total) return;
    int b = idx / Gsz, j = idx - b * Gsz;
    const float* gib = gi + (size_t)b * 3 * Gsz;
    const float* ghb = gh + (size_t)b * 3 * Gsz;
    float r  = 1.0f / (1.0f + expf(-(gib[j] + ghb[j])));
    float z  = 1.0f / (1.0f + expf(-(gib[Gsz + j] + ghb[Gsz + j])));
    float nn = tanhf(gib[2 * Gsz + j] + r * ghb[2 * Gsz + j]);
    float hv = h[idx];
    float out = (1.0f - z) * nn + z * hv;
    h_new[idx] = out;
    h_bf[idx]  = (__bf16)out;
    if (h_bf2) h_bf2[(size_t)b * ld_bf2 + j] = (__bf16)out;
}

// o[r,c] = bf16(a[r,c] + b[r,c]) with independent row strides
__global__ void add2_cvt_kernel(const float* __restrict__ a, int lda,
                                const float* __restrict__ b, int ldb,
                                __bf16* __restrict__ o, int ldo, int rows, int cols) {
    int idx = blockIdx.x * blockDim.x + threadIdx.x;
    if (idx >= rows * cols) return;
    int r = idx / cols, c = idx - r * cols;
    o[(size_t)r * ldo + c] = (__bf16)(a[(size_t)r * lda + c] + b[(size_t)r * ldb + c]);
}

// scores[l,b] = <q[b,:], hist[l,b,:]>   (l = 0..L-1)
__global__ void att_scores_kernel(const float* __restrict__ q, const float* __restrict__ hist,
                                  float* __restrict__ s, int L) {
    int idx = blockIdx.x * blockDim.x + threadIdx.x;
    if (idx >= L * B_DLG) return;
    int l = idx / B_DLG, b = idx - l * B_DLG;
    const float* hp = hist + ((size_t)l * B_DLG + b) * E_UTT;
    const float* qp = q + (size_t)b * E_UTT;
    float acc = 0.0f;
    for (int e = 0; e < E_UTT; ++e) acc += qp[e] * hp[e];
    s[idx] = acc;
}

// in-place softmax over l (per dialogue b); one wave of 32 lanes
__global__ void att_softmax_kernel(float* __restrict__ s, int L) {
    int b = threadIdx.x;
    if (b >= B_DLG) return;
    float mx = -1e30f;
    for (int l = 0; l < L; ++l) mx = fmaxf(mx, s[l * B_DLG + b]);
    float sum = 0.0f;
    for (int l = 0; l < L; ++l) {
        float v = expf(s[l * B_DLG + b] - mx);
        s[l * B_DLG + b] = v;
        sum += v;
    }
    float inv = 1.0f / sum;
    for (int l = 0; l < L; ++l) s[l * B_DLG + b] *= inv;
}

// o_bf[b,e] = bf16( sum_l w[l,b]*hist[l,b,e] + xi[b,e] )
__global__ void att_ctx_kernel(const float* __restrict__ w, const float* __restrict__ hist,
                               const float* __restrict__ xi, int ld_xi,
                               __bf16* __restrict__ o, int L) {
    int idx = blockIdx.x * blockDim.x + threadIdx.x;
    if (idx >= B_DLG * E_UTT) return;
    int b = idx / E_UTT, e = idx - b * E_UTT;
    float sacc = 0.0f;
    for (int l = 0; l < L; ++l)
        sacc += w[l * B_DLG + b] * hist[((size_t)l * B_DLG + b) * E_UTT + e];
    o[idx] = (__bf16)(sacc + xi[(size_t)b * ld_xi + e]);
}

// ---------------------------------------------------------------------------
// Host orchestration
// ---------------------------------------------------------------------------
extern "C" void kernel_launch(void* const* d_in, const int* in_sizes, int n_in,
                              void* d_out, int out_size, void* d_ws, size_t ws_size,
                              hipStream_t stream)
{
    (void)in_sizes; (void)n_in; (void)out_size; (void)ws_size;

    const float* x_in   = (const float*)d_in[0];
    const float* WihF_f = (const float*)d_in[1];
    const float* WhhF_f = (const float*)d_in[2];
    const float* bF     = (const float*)d_in[3];
    const float* WihB_f = (const float*)d_in[4];
    const float* WhhB_f = (const float*)d_in[5];
    const float* bB     = (const float*)d_in[6];
    const float* Watt_f = (const float*)d_in[7];
    const float* gWih_f = (const float*)d_in[8];
    const float* gWhh_f = (const float*)d_in[9];
    const float* g_bih  = (const float*)d_in[10];
    const float* g_bhh  = (const float*)d_in[11];
    const float* pWih_f = (const float*)d_in[12];
    const float* pWhh_f = (const float*)d_in[13];
    const float* p_bih  = (const float*)d_in[14];
    const float* p_bhh  = (const float*)d_in[15];
    const float* eWih_f = (const float*)d_in[16];
    const float* eWhh_f = (const float*)d_in[17];
    const float* e_bih  = (const float*)d_in[18];
    const float* e_bhh  = (const float*)d_in[19];
    const float* clsW_f = (const float*)d_in[20];
    const float* cls_b  = (const float*)d_in[21];

    // ---- workspace carve-out ----
    size_t off = 0;
    auto alloc = [&](size_t bytes) -> void* {
        void* p = (char*)d_ws + off;
        off += (bytes + 255) & ~(size_t)255;
        return p;
    };
    __bf16* Xbf     = (__bf16*)alloc((size_t)N_UTTER * TW_WORDS * EW_PAD * 2);
    __bf16* WihF    = (__bf16*)alloc((size_t)G4H * EW_PAD * 2);
    __bf16* WhhF    = (__bf16*)alloc((size_t)G4H * H_LSTM * 2);
    __bf16* WihB    = (__bf16*)alloc((size_t)G4H * EW_PAD * 2);
    __bf16* WhhB    = (__bf16*)alloc((size_t)G4H * H_LSTM * 2);
    __bf16* WattT   = (__bf16*)alloc((size_t)E_UTT * E_UTT * 2);
    __bf16* gWih    = (__bf16*)alloc((size_t)E3 * E_UTT * 2);
    __bf16* gWhh    = (__bf16*)alloc((size_t)E3 * E_UTT * 2);
    __bf16* pWih    = (__bf16*)alloc((size_t)E3 * E_UTT * 2);
    __bf16* pWhh    = (__bf16*)alloc((size_t)E3 * E_UTT * 2);
    __bf16* eWih    = (__bf16*)alloc((size_t)HH3 * E_UTT * 2);
    __bf16* eWhh    = (__bf16*)alloc((size_t)HH3 * HH_EMO * 2);
    __bf16* clsW    = (__bf16*)alloc((size_t)C_CLS * HH_EMO * 2);
    float*  Gbuf    = (float*)alloc((size_t)N_UTTER * G4H * 4);
    float*  cbuf    = (float*)alloc((size_t)N_UTTER * H_LSTM * 4);
    __bf16* hbf     = (__bf16*)alloc((size_t)N_UTTER * H_LSTM * 2);
    float*  featsf  = (float*)alloc((size_t)N_UTTER * E_UTT * 4);
    __bf16* featsbf = (__bf16*)alloc((size_t)N_UTTER * E_UTT * 2);
    float*  histf   = (float*)alloc((size_t)(T_UTT + 1) * B_DLG * E_UTT * 4);
    __bf16* histbf  = (__bf16*)alloc((size_t)(T_UTT + 1) * B_DLG * E_UTT * 2);
    float*  partyf  = (float*)alloc((size_t)2 * B_DLG * E_UTT * 4);
    __bf16* partybf = (__bf16*)alloc((size_t)2 * B_DLG * E_UTT * 2);
    float*  emof    = (float*)alloc((size_t)B_DLG * HH_EMO * 4);
    __bf16* emobf   = (__bf16*)alloc((size_t)B_DLG * HH_EMO * 2);
    __bf16* emoseq  = (__bf16*)alloc((size_t)B_DLG * T_UTT * HH_EMO * 2);
    float*  qbuf    = (float*)alloc((size_t)B_DLG * E_UTT * 4);
    float*  scores  = (float*)alloc((size_t)(T_UTT + 1) * B_DLG * 4);
    __bf16* xgbf    = (__bf16*)alloc((size_t)B_DLG * E_UTT * 2);
    __bf16* xpbf    = (__bf16*)alloc((size_t)B_DLG * E_UTT * 2);
    float*  Gi      = (float*)alloc((size_t)B_DLG * E3 * 4);
    float*  Gh      = (float*)alloc((size_t)B_DLG * E3 * 4);

    auto cdiv = [](long a, long b) { return (int)((a + b - 1) / b); };
    auto cvt = [&](const float* src, __bf16* dst, int R, int Cc, int Cp) {
        cvt_pad_kernel<<<cdiv((long)R * Cp, 256), 256, 0, stream>>>(src, dst, R, Cc, Cp);
    };
    auto gemm = [&](const __bf16* A1, int lda1, const __bf16* B1, int ldb1, int K1,
                    const __bf16* A2, int lda2, const __bf16* B2, int ldb2, int K2,
                    const float* bias, float* C, int ldc, int M, int N) {
        dim3 grid((N + 63) / 64, (M + 127) / 128);
        gemm_bf16_wmma<<<grid, 256, 0, stream>>>(A1, lda1, B1, ldb1, K1,
                                                 A2, lda2, B2, ldb2, K2,
                                                 bias, C, ldc, M, N);
    };

    // ---- weight / input conversion to bf16 (K zero-padded) ----
    cvt(x_in,   Xbf,  N_UTTER * TW_WORDS, EW_IN, EW_PAD);
    cvt(WihF_f, WihF, G4H, EW_IN, EW_PAD);
    cvt(WhhF_f, WhhF, G4H, H_LSTM, H_LSTM);
    cvt(WihB_f, WihB, G4H, EW_IN, EW_PAD);
    cvt(WhhB_f, WhhB, G4H, H_LSTM, H_LSTM);
    cvt_transpose_kernel<<<cdiv((long)E_UTT * E_UTT, 256), 256, 0, stream>>>(Watt_f, WattT, E_UTT, E_UTT);
    cvt(gWih_f, gWih, E3, E_UTT, E_UTT);
    cvt(gWhh_f, gWhh, E3, E_UTT, E_UTT);
    cvt(pWih_f, pWih, E3, E_UTT, E_UTT);
    cvt(pWhh_f, pWhh, E3, E_UTT, E_UTT);
    cvt(eWih_f, eWih, HH3, E_UTT, E_UTT);
    cvt(eWhh_f, eWhh, HH3, HH_EMO, HH_EMO);
    cvt(clsW_f, clsW, C_CLS, HH_EMO, HH_EMO);

    const int NH  = N_UTTER * H_LSTM;
    const int ldX = TW_WORDS * EW_PAD;   // row stride of one word-timestep slice

    // ---- Phase 1: BiLSTM over words (forward direction) ----
    hipMemsetAsync(cbuf, 0, (size_t)NH * 4, stream);
    for (int t = 0; t < TW_WORDS; ++t) {
        gemm(Xbf + (size_t)t * EW_PAD, ldX, WihF, EW_PAD, EW_PAD,
             (t == 0) ? nullptr : hbf, H_LSTM, WhhF, H_LSTM, H_LSTM,
             bF, Gbuf, G4H, N_UTTER, G4H);
        lstm_gate_kernel<<<cdiv(NH, 256), 256, 0, stream>>>(
            Gbuf, cbuf, cbuf, hbf,
            (t == TW_WORDS - 1) ? featsf : nullptr, E_UTT, NH);
    }
    // ---- backward direction (reversed word order) ----
    hipMemsetAsync(cbuf, 0, (size_t)NH * 4, stream);
    for (int s = 0; s < TW_WORDS; ++s) {
        const int t = TW_WORDS - 1 - s;
        gemm(Xbf + (size_t)t * EW_PAD, ldX, WihB, EW_PAD, EW_PAD,
             (s == 0) ? nullptr : hbf, H_LSTM, WhhB, H_LSTM, H_LSTM,
             bB, Gbuf, G4H, N_UTTER, G4H);
        lstm_gate_kernel<<<cdiv(NH, 256), 256, 0, stream>>>(
            Gbuf, cbuf, cbuf, hbf,
            (s == TW_WORDS - 1) ? (featsf + H_LSTM) : nullptr, E_UTT, NH);
    }
    // feats f32 -> bf16 (phase-2 GEMM A operands)
    cvt(featsf, featsbf, N_UTTER, E_UTT, E_UTT);

    // ---- Phase 2: DialogueRNN recurrence over utterances ----
    const size_t BE = (size_t)B_DLG * E_UTT;     // 32768
    const size_t BH = (size_t)B_DLG * HH_EMO;    // 16384
    hipMemsetAsync(histf,   0, BE * 4, stream);           // hist slot 0
    hipMemsetAsync(histbf,  0, BE * 2, stream);
    hipMemsetAsync(partyf,  0, 2 * BE * 4, stream);
    hipMemsetAsync(partybf, 0, 2 * BE * 2, stream);
    hipMemsetAsync(emof,    0, BH * 4, stream);
    hipMemsetAsync(emobf,   0, BH * 2, stream);

    const int ldFeat = T_UTT * E_UTT;            // row stride across dialogues (fixed i)
    for (int i = 0; i < T_UTT; ++i) {
        const int p = i & 1;
        const float*  xi_f  = featsf  + (size_t)i * E_UTT;
        const __bf16* xi_bf = featsbf + (size_t)i * E_UTT;
        const int L = i + 1;

        // xg = bf16(x_i + party[p])
        add2_cvt_kernel<<<cdiv((long)BE, 256), 256, 0, stream>>>(
            xi_f, ldFeat, partyf + (size_t)p * BE, E_UTT, xgbf, E_UTT, B_DLG, E_UTT);
        // global-state GRU
        gemm(xgbf, E_UTT, gWih, E_UTT, E_UTT, nullptr, 0, nullptr, 0, 0,
             g_bih, Gi, E3, B_DLG, E3);
        gemm(histbf + (size_t)i * BE, E_UTT, gWhh, E_UTT, E_UTT, nullptr, 0, nullptr, 0, 0,
             g_bhh, Gh, E3, B_DLG, E3);
        gru_gate_kernel<<<cdiv((long)BE, 256), 256, 0, stream>>>(
            Gi, Gh, histf + (size_t)i * BE, histf + (size_t)(i + 1) * BE,
            histbf + (size_t)(i + 1) * BE, nullptr, 0, E_UTT, (int)BE);

        // attention: q = x_i @ W_att ; masked softmax over hist[0..i]; context
        gemm(xi_bf, ldFeat, WattT, E_UTT, E_UTT, nullptr, 0, nullptr, 0, 0,
             nullptr, qbuf, E_UTT, B_DLG, E_UTT);
        att_scores_kernel<<<cdiv((long)L * B_DLG, 256), 256, 0, stream>>>(qbuf, histf, scores, L);
        att_softmax_kernel<<<1, 32, 0, stream>>>(scores, L);
        att_ctx_kernel<<<cdiv((long)BE, 256), 256, 0, stream>>>(scores, histf, xi_f, ldFeat, xpbf, L);

        // party-state GRU
        gemm(xpbf, E_UTT, pWih, E_UTT, E_UTT, nullptr, 0, nullptr, 0, 0,
             p_bih, Gi, E3, B_DLG, E3);
        gemm(partybf + (size_t)p * BE, E_UTT, pWhh, E_UTT, E_UTT, nullptr, 0, nullptr, 0, 0,
             p_bhh, Gh, E3, B_DLG, E3);
        gru_gate_kernel<<<cdiv((long)BE, 256), 256, 0, stream>>>(
            Gi, Gh, partyf + (size_t)p * BE, partyf + (size_t)p * BE,
            partybf + (size_t)p * BE, nullptr, 0, E_UTT, (int)BE);

        // emotion-state GRU (input = new_p); also writes emo_seq bf16 slice
        gemm(partybf + (size_t)p * BE, E_UTT, eWih, E_UTT, E_UTT, nullptr, 0, nullptr, 0, 0,
             e_bih, Gi, HH3, B_DLG, HH3);
        gemm(emobf, HH_EMO, eWhh, HH_EMO, HH_EMO, nullptr, 0, nullptr, 0, 0,
             e_bhh, Gh, HH3, B_DLG, HH3);
        gru_gate_kernel<<<cdiv((long)BH, 256), 256, 0, stream>>>(
            Gi, Gh, emof, emof, emobf,
            emoseq + (size_t)i * HH_EMO, T_UTT * HH_EMO, HH_EMO, (int)BH);
    }

    // ---- classifier: out[B*T, C] = emo_seq @ cls_W^T + cls_b ----
    gemm(emoseq, HH_EMO, clsW, HH_EMO, HH_EMO, nullptr, 0, nullptr, 0, 0,
         cls_b, (float*)d_out, C_CLS, N_UTTER, C_CLS);
}
